// LinearTransformer_48919677501974
// MI455X (gfx1250) — matrix operand order repaired and checked by reading, not compiled, not run on hardware
//
#include <hip/hip_runtime.h>

// ---------------- model dims ----------------
#define BB   4
#define LL   2048
#define HH   12
#define DHD  64
#define DM   768
#define DFF  3072
#define NLAYER 12
#define MTOT (BB*LL)          // 8192 tokens

// ---------------- types ----------------
typedef __attribute__((ext_vector_type(16))) __bf16          v16bf;
typedef __attribute__((ext_vector_type(8)))  float           v8f;
typedef __attribute__((ext_vector_type(8)))  unsigned short  u16x8;   // 16 bytes
typedef __attribute__((ext_vector_type(4)))  float           f32x4;   // 16 bytes
typedef __attribute__((ext_vector_type(4)))  int             v4i;     // 16 bytes

union Frag16 {
  v16bf v;
  u16x8 h[2];
};

// gfx1250 async global->LDS copy engine (ASYNCcnt), if this toolchain exposes it.
#if defined(__has_builtin)
# if __has_builtin(__builtin_amdgcn_global_load_async_to_lds_b128) && \
     __has_builtin(__builtin_amdgcn_s_wait_asynccnt)
#  define USE_ASYNC_LDS 1
# endif
#endif
#ifndef USE_ASYNC_LDS
# define USE_ASYNC_LDS 0
#endif

#if USE_ASYNC_LDS
typedef __attribute__((address_space(1))) v4i* gptr_b128;   // global (__device__) int4*
typedef __attribute__((address_space(3))) v4i* sptr_b128;   // LDS (__shared__) int4*
#endif

__device__ __forceinline__ unsigned short f32_to_bf16u(float f) {
  unsigned u = __float_as_uint(f);
  unsigned r = u + 0x7FFFu + ((u >> 16) & 1u);   // round-to-nearest-even
  return (unsigned short)(r >> 16);
}

// =====================================================================
// GEMM: C[M,N] = A_bf16[M,K] * W_f32[N,K]^T + bias, fused epilogue.
// ACT: 0 = none, 1 = elu+1, 2 = exact gelu
// Block tile 128(M) x 64(N), 8 waves of 32, each wave 32x32 (2x2 WMMA).
// =====================================================================
template<int ACT>
__global__ __launch_bounds__(256) void gemm_bf16_wmma(
    const unsigned short* __restrict__ A,   // [M,K] bf16 row-major
    const float*          __restrict__ W,   // [N,K] f32 row-major
    const float*          __restrict__ bias,// [N] or null
    const float*          __restrict__ rowscale, // [M] or null (attention mask)
    float*                __restrict__ outF, // [M,N] f32 or null
    unsigned short*       __restrict__ outB, // [M,N] bf16 or null
    int M, int N, int K)
{
  __shared__ __align__(16) unsigned short As[128][32];
  __shared__ __align__(16) unsigned short Bs[64][32];

  const int t    = threadIdx.x;
  const int lane = t & 31;
  const int wv   = t >> 5;
  const int wm   = (wv & 3) * 32;   // wave M offset in tile
  const int wn   = (wv >> 2) * 32;  // wave N offset in tile
  const int M0   = blockIdx.y * 128;
  const int N0   = blockIdx.x * 64;

  const int fr = lane & 15;
  const int hi = lane >> 4;

  v8f acc[2][2] = {};

  const int arow = t >> 1, ahalf = t & 1;   // A tile: 128 rows x 32 cols bf16
  const int brow = t >> 2, bq    = t & 3;   // B tile:  64 rows x 32 cols f32->bf16

  for (int kb = 0; kb < K; kb += 32) {
    const unsigned short* agsrc = A + (size_t)(M0 + arow) * K + kb + 16 * ahalf;

    // ---- B tile: global f32 load + on-the-fly bf16 convert (VGPR path) ----
    const f32x4* bsrc = (const f32x4*)(W + (size_t)(N0 + brow) * K + kb + 8 * bq);
    f32x4 b0 = bsrc[0];
    f32x4 b1 = bsrc[1];
    u16x8 bh;
    bh[0] = f32_to_bf16u(b0[0]); bh[1] = f32_to_bf16u(b0[1]);
    bh[2] = f32_to_bf16u(b0[2]); bh[3] = f32_to_bf16u(b0[3]);
    bh[4] = f32_to_bf16u(b1[0]); bh[5] = f32_to_bf16u(b1[1]);
    bh[6] = f32_to_bf16u(b1[2]); bh[7] = f32_to_bf16u(b1[3]);

#if !USE_ASYNC_LDS
    const u16x8* asrc = (const u16x8*)agsrc;
    u16x8 a0 = asrc[0];
    u16x8 a1 = asrc[1];
#endif

    if (kb + 32 < K) {   // hint next K tile into cache (global_prefetch_b8)
      __builtin_prefetch(agsrc + 32, 0, 3);
      __builtin_prefetch(W + (size_t)(N0 + brow) * K + kb + 32 + 8 * bq, 0, 3);
    }

    __syncthreads();     // previous iteration's LDS readers are done

#if USE_ASYNC_LDS
    // A tile: async copy global -> LDS, 2 x b128 per thread (ASYNCcnt path)
    __builtin_amdgcn_global_load_async_to_lds_b128(
        (gptr_b128)agsrc, (sptr_b128)&As[arow][16 * ahalf], 0, 0);
    __builtin_amdgcn_global_load_async_to_lds_b128(
        (gptr_b128)(agsrc + 8), (sptr_b128)&As[arow][16 * ahalf + 8], 0, 0);
#else
    *(u16x8*)&As[arow][16 * ahalf]     = a0;
    *(u16x8*)&As[arow][16 * ahalf + 8] = a1;
#endif
    *(u16x8*)&Bs[brow][8 * bq] = bh;

#if USE_ASYNC_LDS
    __builtin_amdgcn_s_wait_asynccnt(0);
#endif
    __syncthreads();

    // ---- fragments (CDNA5 16-bit WMMA VGPR layout) ----
    // A: lane(l): row = fr, elems 0..7 -> K = 8*hi+j ; elems 8..15 -> K = 16+8*hi+j
    // B: lane(l): col = fr, elems 0..7 -> K = 8*hi+j ; elems 8..15 -> K = 16+8*hi+j
    Frag16 fa[2], fb[2];
    #pragma unroll
    for (int mi = 0; mi < 2; ++mi) {
      fa[mi].h[0] = *(const u16x8*)&As[wm + 16 * mi + fr][8 * hi];
      fa[mi].h[1] = *(const u16x8*)&As[wm + 16 * mi + fr][16 + 8 * hi];
    }
    #pragma unroll
    for (int ni = 0; ni < 2; ++ni) {
      fb[ni].h[0] = *(const u16x8*)&Bs[wn + 16 * ni + fr][8 * hi];
      fb[ni].h[1] = *(const u16x8*)&Bs[wn + 16 * ni + fr][16 + 8 * hi];
    }
    #pragma unroll
    for (int mi = 0; mi < 2; ++mi)
      #pragma unroll
      for (int ni = 0; ni < 2; ++ni)
        acc[mi][ni] = __builtin_amdgcn_wmma_f32_16x16x32_bf16(
            false, fa[mi].v, false, fb[ni].v, (short)0, acc[mi][ni], false, false);
  }

  // ---- epilogue: D layout -> lane(l): col = fr; VGPR r: row = 16*mi + 8*hi + r
  #pragma unroll
  for (int mi = 0; mi < 2; ++mi) {
    #pragma unroll
    for (int ni = 0; ni < 2; ++ni) {
      int gn = N0 + wn + 16 * ni + fr;
      float bv = bias ? bias[gn] : 0.f;
      #pragma unroll
      for (int r = 0; r < 8; ++r) {
        int gm = M0 + wm + 16 * mi + 8 * hi + r;
        float x = acc[mi][ni][r] + bv;
        if (ACT == 1) x = (x > 0.f) ? (x + 1.f) : expf(x);      // elu(x)+1
        if (ACT == 2) x = 0.5f * x * (1.f + erff(x * 0.70710678118654752f));
        if (rowscale) x *= rowscale[gm];
        size_t o = (size_t)gm * N + gn;
        if (outF) outF[o] = x;
        if (outB) outB[o] = f32_to_bf16u(x);
      }
    }
  }
}

// =====================================================================
// Embedding gather + LayerNorm (eps 1e-12) -> x f32, x bf16
// =====================================================================
__global__ __launch_bounds__(256) void k_embed_ln(
    const int* __restrict__ ids, const int* __restrict__ tti,
    const float* __restrict__ we, const float* __restrict__ pe, const float* __restrict__ te,
    const float* __restrict__ lw, const float* __restrict__ lb,
    float* __restrict__ xf, unsigned short* __restrict__ xb)
{
  int i = blockIdx.x;            // token
  int l = i % LL;
  int t = threadIdx.x;
  int id = ids[i];
  int tt = tti[i];
  float e[3], s = 0.f, q = 0.f;
  #pragma unroll
  for (int j = 0; j < 3; ++j) {
    int c = t + j * 256;
    float v = we[(size_t)id * DM + c] + pe[(size_t)l * DM + c] + te[(size_t)tt * DM + c];
    e[j] = v; s += v; q += v * v;
  }
  __shared__ float s1[256], s2[256];
  s1[t] = s; s2[t] = q; __syncthreads();
  for (int o = 128; o > 0; o >>= 1) {
    if (t < o) { s1[t] += s1[t + o]; s2[t] += s2[t + o]; }
    __syncthreads();
  }
  float mean = s1[0] * (1.f / DM);
  float var  = s2[0] * (1.f / DM) - mean * mean;
  float inv  = rsqrtf(var + 1e-12f);
  #pragma unroll
  for (int j = 0; j < 3; ++j) {
    int c = t + j * 256;
    float v = (e[j] - mean) * inv * lw[c] + lb[c];
    xf[(size_t)i * DM + c] = v;
    xb[(size_t)i * DM + c] = f32_to_bf16u(v);
  }
}

// =====================================================================
// Residual add + LayerNorm (eps 1e-5) -> out f32, out bf16
// =====================================================================
__global__ __launch_bounds__(256) void k_add_ln(
    const float* __restrict__ x, const float* __restrict__ d,
    const float* __restrict__ lw, const float* __restrict__ lb, float eps,
    float* __restrict__ outF, unsigned short* __restrict__ outB)
{
  int i = blockIdx.x;
  int t = threadIdx.x;
  float e[3], s = 0.f, q = 0.f;
  #pragma unroll
  for (int j = 0; j < 3; ++j) {
    int c = t + j * 256;
    float v = x[(size_t)i * DM + c] + d[(size_t)i * DM + c];
    e[j] = v; s += v; q += v * v;
  }
  __shared__ float s1[256], s2[256];
  s1[t] = s; s2[t] = q; __syncthreads();
  for (int o = 128; o > 0; o >>= 1) {
    if (t < o) { s1[t] += s1[t + o]; s2[t] += s2[t + o]; }
    __syncthreads();
  }
  float mean = s1[0] * (1.f / DM);
  float var  = s2[0] * (1.f / DM) - mean * mean;
  float inv  = rsqrtf(var + eps);
  #pragma unroll
  for (int j = 0; j < 3; ++j) {
    int c = t + j * 256;
    float v = (e[j] - mean) * inv * lw[c] + lb[c];
    outF[(size_t)i * DM + c] = v;
    outB[(size_t)i * DM + c] = f32_to_bf16u(v);
  }
}

// =====================================================================
// KV[b,h,m,d] = sum_l K[b,l,h,d] * V[b,l,h,m]  ;  Ksum[b,h,d] = sum_l K
// One block per (b,h); 256 threads; thread t owns m = t/4, d in [(t%4)*16, +16)
// =====================================================================
__global__ __launch_bounds__(256) void k_kv(
    const float* __restrict__ kk, const float* __restrict__ vv,
    float* __restrict__ KV, float* __restrict__ Ksum)
{
  int bh = blockIdx.x;
  int b = bh / HH, h = bh % HH;
  int t = threadIdx.x;
  __shared__ float Kl[DHD], Vl[DHD];
  float acc[16];
  #pragma unroll
  for (int j = 0; j < 16; ++j) acc[j] = 0.f;
  float ks = 0.f;
  int m = t >> 2, db = (t & 3) * 16;
  for (int l = 0; l < LL; ++l) {
    size_t base = ((size_t)(b * LL + l)) * DM + h * DHD;
    if (t < DHD)            Kl[t]       = kk[base + t];
    else if (t < 2 * DHD)   Vl[t - DHD] = vv[base + (t - DHD)];
    __syncthreads();
    float vm = Vl[m];
    #pragma unroll
    for (int j = 0; j < 16; ++j) acc[j] += Kl[db + j] * vm;
    if (t < DHD) ks += Kl[t];
    __syncthreads();
  }
  #pragma unroll
  for (int j = 0; j < 16; ++j)
    KV[(size_t)bh * DHD * DHD + m * DHD + db + j] = acc[j];
  if (t < DHD) Ksum[bh * DHD + t] = ks;
}

// =====================================================================
// a[b,l,h,m] = (sum_d Q[d]*KV[m,d]) / (sum_d Q[d]*Ksum[d] + eps) -> bf16
// One block per (token, head); 64 threads, thread = m.
// =====================================================================
__global__ __launch_bounds__(64) void k_attn_out(
    const float* __restrict__ q, const float* __restrict__ KV,
    const float* __restrict__ Ksum, unsigned short* __restrict__ ab)
{
  int i = blockIdx.x;          // token
  int h = blockIdx.y;
  int b = i / LL;
  int bh = b * HH + h;
  int t = threadIdx.x;         // m
  __shared__ float Qs[DHD];
  Qs[t] = q[(size_t)i * DM + h * DHD + t];
  __syncthreads();
  float s = 0.f;
  #pragma unroll 8
  for (int dd = 0; dd < DHD; ++dd) s += Qs[dd] * Ksum[bh * DHD + dd];
  float Z = 1.f / (s + 1e-6f);
  float a = 0.f;
  const float* kvrow = KV + (size_t)bh * DHD * DHD + t * DHD;
  #pragma unroll 8
  for (int dd = 0; dd < DHD; ++dd) a += Qs[dd] * kvrow[dd];
  ab[(size_t)i * DM + h * DHD + t] = f32_to_bf16u(a * Z);
}

// =====================================================================
// Host orchestration
// =====================================================================
extern "C" void kernel_launch(void* const* d_in, const int* in_sizes, int n_in,
                              void* d_out, int out_size, void* d_ws, size_t ws_size,
                              hipStream_t stream) {
  const int*   ids  = (const int*)d_in[0];
  const float* mask = (const float*)d_in[1];
  const int*   tti  = (const int*)d_in[2];
  const float* we   = (const float*)d_in[3];
  const float* pe   = (const float*)d_in[4];
  const float* te   = (const float*)d_in[5];
  const float* elw  = (const float*)d_in[6];
  const float* elb  = (const float*)d_in[7];
  const float* Wq = (const float*)d_in[8];   const float* bq = (const float*)d_in[9];
  const float* Wk = (const float*)d_in[10];  const float* bk = (const float*)d_in[11];
  const float* Wv = (const float*)d_in[12];  const float* bv = (const float*)d_in[13];
  const float* Wo = (const float*)d_in[14];  const float* bo = (const float*)d_in[15];
  const float* l1w = (const float*)d_in[16]; const float* l1b = (const float*)d_in[17];
  const float* W1 = (const float*)d_in[18];  const float* b1 = (const float*)d_in[19];
  const float* W2 = (const float*)d_in[20];  const float* b2 = (const float*)d_in[21];
  const float* l2w = (const float*)d_in[22]; const float* l2b = (const float*)d_in[23];

  char* ws = (char*)d_ws;
  size_t off = 0;
  auto grab = [&](size_t bytes) -> void* {
    void* p = ws + off;
    off += (bytes + 255) & ~(size_t)255;
    return p;
  };
  float*          xf  = (float*)grab((size_t)MTOT * DM * 4);
  unsigned short* xb  = (unsigned short*)grab((size_t)MTOT * DM * 2);
  float*          qf  = (float*)grab((size_t)MTOT * DM * 4);
  float*          kf  = (float*)grab((size_t)MTOT * DM * 4);
  float*          vf  = (float*)grab((size_t)MTOT * DM * 4);
  unsigned short* ab  = (unsigned short*)grab((size_t)MTOT * DM * 2);
  float*          of  = (float*)grab((size_t)MTOT * DM * 4);
  float*          KV  = (float*)grab((size_t)BB * HH * DHD * DHD * 4);
  float*          Ks  = (float*)grab((size_t)BB * HH * DHD * 4);
  unsigned short* y1b = (unsigned short*)grab((size_t)MTOT * DFF * 2);
  float*          y2f = (float*)grab((size_t)MTOT * DM * 4);

  k_embed_ln<<<MTOT, 256, 0, stream>>>(ids, tti, we, pe, te, elw, elb, xf, xb);

  dim3 g768(DM / 64, MTOT / 128);
  dim3 gFF(DFF / 64, MTOT / 128);

  for (int Li = 0; Li < NLAYER; ++Li) {
    const float* wq = Wq + (size_t)Li * DM * DM;
    const float* wk = Wk + (size_t)Li * DM * DM;
    const float* wv = Wv + (size_t)Li * DM * DM;
    const float* wo = Wo + (size_t)Li * DM * DM;
    const float* w1 = W1 + (size_t)Li * DFF * DM;
    const float* w2 = W2 + (size_t)Li * DM * DFF;

    // Q = elu(xWq^T+bq)+1 ; K = (elu(xWk^T+bk)+1)*mask ; V = xWv^T+bv
    gemm_bf16_wmma<1><<<g768, 256, 0, stream>>>(xb, wq, bq + Li * DM, nullptr,
                                                qf, nullptr, MTOT, DM, DM);
    gemm_bf16_wmma<1><<<g768, 256, 0, stream>>>(xb, wk, bk + Li * DM, mask,
                                                kf, nullptr, MTOT, DM, DM);
    gemm_bf16_wmma<0><<<g768, 256, 0, stream>>>(xb, wv, bv + Li * DM, nullptr,
                                                vf, nullptr, MTOT, DM, DM);

    k_kv<<<BB * HH, 256, 0, stream>>>(kf, vf, KV, Ks);
    k_attn_out<<<dim3(MTOT, HH), 64, 0, stream>>>(qf, KV, Ks, ab);

    // O-projection, then residual + LN1
    gemm_bf16_wmma<0><<<g768, 256, 0, stream>>>(ab, wo, bo + Li * DM, nullptr,
                                                of, nullptr, MTOT, DM, DM);
    k_add_ln<<<MTOT, 256, 0, stream>>>(xf, of, l1w + Li * DM, l1b + Li * DM, 1e-5f,
                                       xf, xb);

    // FFN: gelu(x W1^T + b1) W2^T + b2, then residual + LN2
    gemm_bf16_wmma<2><<<gFF, 256, 0, stream>>>(xb, w1, b1 + Li * DFF, nullptr,
                                               nullptr, y1b, MTOT, DFF, DM);
    gemm_bf16_wmma<0><<<g768, 256, 0, stream>>>(y1b, w2, b2 + Li * DM, nullptr,
                                                y2f, nullptr, MTOT, DM, DFF);

    float* xdst = (Li == NLAYER - 1) ? (float*)d_out : xf;
    k_add_ln<<<MTOT, 256, 0, stream>>>(xf, y2f, l2w + Li * DM, l2b + Li * DM, 1e-5f,
                                       xdst, xb);
  }
}
